// Edges_14929306321382
// MI455X (gfx1250) — matrix operand order repaired and checked by reading, not compiled, not run on hardware
//
#include <hip/hip_runtime.h>

// ---- problem constants (from reference) ----
#define FEAT      64
#define HIDDEN    128
#define OUTD      64
#define NMU       100
#define K1        228          // 2*FEAT + NMU
#define K1P       256          // padded K for GEMM1
#define TILE_E    256          // edges per block (8 waves x 32)

// ---- LDS strides (halves), padded so row step = 4 banks (conflict-free) ----
#define W1_STRIDE 264          // 132 dwords -> +4 banks/row
#define W2_STRIDE 136          // 68 dwords  -> +4 banks/row
#define HDN_STRIDE 136

#define LDS_W1_HALFS  (HIDDEN * W1_STRIDE)        // 33792
#define LDS_W2_HALFS  (OUTD   * W2_STRIDE)        // 8704
#define LDS_HDN_HALFS (8 * 32 * HDN_STRIDE)       // 34816 (32 hdn rows per wave)
#define LDS_BYTES ((LDS_W1_HALFS + LDS_W2_HALFS + LDS_HDN_HALFS) * 2 + 128 * 4)

#define LOG2E      1.4426950408889634f
#define C_RBF     (-10.0f * LOG2E)                // exp(-10 t^2) = exp2(C_RBF * t^2)

typedef _Float16 v16h __attribute__((ext_vector_type(16)));
typedef _Float16 v8h  __attribute__((ext_vector_type(8)));
typedef float    v8f  __attribute__((ext_vector_type(8)));
typedef float    v4f  __attribute__((ext_vector_type(4)));

union AB16 { v16h v; v8h h[2]; };

// load 8 contiguous f32 from global, convert to 8 halves
__device__ inline v8h cvt8_f32(const float* __restrict__ p) {
  v4f a = *(const v4f*)p;
  v4f b = *(const v4f*)(p + 4);
  v8h r;
  r[0] = (_Float16)a[0]; r[1] = (_Float16)a[1];
  r[2] = (_Float16)a[2]; r[3] = (_Float16)a[3];
  r[4] = (_Float16)b[0]; r[5] = (_Float16)b[1];
  r[6] = (_Float16)b[2]; r[7] = (_Float16)b[3];
  return r;
}

// assemble a 16-half fragment from two 16B-aligned LDS groups
__device__ inline v16h ld_frag(const _Float16* p0, const _Float16* p1) {
  AB16 u;
  u.h[0] = *(const v8h*)p0;
  u.h[1] = *(const v8h*)p1;
  return u.v;
}

__global__ __launch_bounds__(256)
void edge_mlp_wmma(const float* __restrict__ h,
                   const int*   __restrict__ src,
                   const int*   __restrict__ dst,
                   const float* __restrict__ enorm,
                   const float* __restrict__ mu,
                   const float* __restrict__ W1,
                   const float* __restrict__ W2,
                   float* __restrict__ out,
                   int E) {
  extern __shared__ char smem[];
  _Float16* sW1  = (_Float16*)smem;                 // [128][264] f16, zero-padded K
  _Float16* sW2  = sW1 + LDS_W1_HALFS;              // [64][136]  f16
  _Float16* sHdn = sW2 + LDS_W2_HALFS;              // 8 waves x [32][136] f16
  float*    sMu  = (float*)(sHdn + LDS_HDN_HALFS);  // [128] f32

  const int tid = threadIdx.x;

  // ---- stage weights into LDS as f16 (zero-pad W1 cols 228..255) ----
  #pragma unroll 4
  for (int i = tid; i < HIDDEN * K1P; i += 256) {
    int r = i >> 8, c = i & (K1P - 1);
    float v = (c < K1) ? W1[r * K1 + c] : 0.0f;
    sW1[r * W1_STRIDE + c] = (_Float16)v;
  }
  #pragma unroll 2
  for (int i = tid; i < OUTD * HIDDEN; i += 256) {
    int r = i >> 7, c = i & (HIDDEN - 1);
    sW2[r * W2_STRIDE + c] = (_Float16)W2[i];
  }
  if (tid < NMU) sMu[tid] = mu[tid];
  __syncthreads();

  const int wave = tid >> 5;
  const int lane = tid & 31;
  const int row  = lane & 15;        // A-matrix row this lane serves
  const int hiK  = lane >> 4;        // lanes 16-31 hold the high K-subgroup
  const int klo  = hiK * 8;

  const int tileBase = blockIdx.x * TILE_E + wave * 32;   // 32 edges per wave

  // ---- A fragments for both 16-edge subtiles -------------------------------
  // A layout (16x32 f16): elems 0..7 = K = c*32+klo.., elems 8..15 = K = +16+klo..
  v16h A[2][8];
  float dval[2];
  #pragma unroll
  for (int s = 0; s < 2; ++s) {
    int edge = tileBase + s * 16 + row;
    if (edge >= E) edge = E - 1;     // clamp (grid divides exactly for E=800000)
    const int   si = src[edge];
    const int   di = dst[edge];
    dval[s] = enorm[edge];
    const float* hs = h + (size_t)si * FEAT;
    const float* hd = h + (size_t)di * FEAT;
    #pragma unroll
    for (int c = 0; c < 4; ++c) {
      const float* base = ((c < 2) ? hs : hd) + (c & 1) * 32;
      AB16 u;
      u.h[0] = cvt8_f32(base + klo);
      u.h[1] = cvt8_f32(base + 16 + klo);
      A[s][c] = u.v;
    }
    const float d = dval[s];
    #pragma unroll
    for (int c = 0; c < 4; ++c) {
      AB16 u;
      #pragma unroll
      for (int g = 0; g < 2; ++g) {
        v8h r;
        #pragma unroll
        for (int j = 0; j < 8; ++j) {
          int mi = c * 32 + g * 16 + klo + j;   // mu index = K - 128
          float val = 0.0f;
          if (mi < NMU) {
            float t = sMu[mi] - d;
            val = __builtin_amdgcn_exp2f(C_RBF * (t * t));
          }
          r[j] = (_Float16)val;
        }
        u.h[g] = r;
      }
      A[s][4 + c] = u.v;
    }
  }

  _Float16* myHdn = sHdn + wave * 32 * HDN_STRIDE;

  // ---- GEMM1: hdn = silu(x @ W1^T) -> [32 x 128] f16 staged in LDS ----
  // B layout (32x16 f16): lane = column n, elems 0..7 = K = c*32 + 8*hiK + 0..7, elems 8..15 = K = c*32 + 16 + 8*hiK + 0..7
  #pragma unroll
  for (int nt = 0; nt < 8; ++nt) {
    v8f acc0 = {}, acc1 = {};
    const _Float16* wrow = sW1 + (nt * 16 + row) * W1_STRIDE + 8 * hiK;
    #pragma unroll
    for (int c = 0; c < 8; ++c) {
      v16h b = ld_frag(wrow + c * 32, wrow + c * 32 + 16);
      acc0 = __builtin_amdgcn_wmma_f32_16x16x32_f16(false, A[0][c], false, b,
                                                    (short)0, acc0, false, false);
      acc1 = __builtin_amdgcn_wmma_f32_16x16x32_f16(false, A[1][c], false, b,
                                                    (short)0, acc1, false, false);
    }
    // C layout: VGPR v -> M = v + 8*hiK, N = nt*16 + row ; SiLU via exp2 + rcp
    #pragma unroll
    for (int v = 0; v < 8; ++v) {
      float x0 = acc0[v];
      float x1 = acc1[v];
      float s0 = x0 * __builtin_amdgcn_rcpf(1.0f + __builtin_amdgcn_exp2f(-LOG2E * x0));
      float s1 = x1 * __builtin_amdgcn_rcpf(1.0f + __builtin_amdgcn_exp2f(-LOG2E * x1));
      myHdn[(v + 8 * hiK) * HDN_STRIDE + nt * 16 + row] = (_Float16)s0;
      myHdn[(16 + v + 8 * hiK) * HDN_STRIDE + nt * 16 + row] = (_Float16)s1;
    }
  }
  __syncthreads();   // LDS transpose visibility (uniform across block)

  // ---- GEMM2: out = hdn @ W2^T -> [32 x 64] ----
  v16h A2[2][4];
  #pragma unroll
  for (int s = 0; s < 2; ++s) {
    #pragma unroll
    for (int c = 0; c < 4; ++c) {
      const _Float16* ap = myHdn + (s * 16 + row) * HDN_STRIDE + c * 32 + klo;
      A2[s][c] = ld_frag(ap, ap + 16);
    }
  }
  #pragma unroll
  for (int nt = 0; nt < 4; ++nt) {
    v8f acc0 = {}, acc1 = {};
    const _Float16* wrow = sW2 + (nt * 16 + row) * W2_STRIDE + 8 * hiK;
    #pragma unroll
    for (int c = 0; c < 4; ++c) {
      v16h b = ld_frag(wrow + c * 32, wrow + c * 32 + 16);
      acc0 = __builtin_amdgcn_wmma_f32_16x16x32_f16(false, A2[0][c], false, b,
                                                    (short)0, acc0, false, false);
      acc1 = __builtin_amdgcn_wmma_f32_16x16x32_f16(false, A2[1][c], false, b,
                                                    (short)0, acc1, false, false);
    }
    #pragma unroll
    for (int v = 0; v < 8; ++v) {
      int e0 = tileBase + v + 8 * hiK;
      int e1 = e0 + 16;
      if (e0 < E) out[(size_t)e0 * OUTD + nt * 16 + row] = acc0[v];
      if (e1 < E) out[(size_t)e1 * OUTD + nt * 16 + row] = acc1[v];
    }
  }
}

extern "C" void kernel_launch(void* const* d_in, const int* in_sizes, int n_in,
                              void* d_out, int out_size, void* d_ws, size_t ws_size,
                              hipStream_t stream) {
  (void)n_in; (void)out_size; (void)d_ws; (void)ws_size;
  const float* h     = (const float*)d_in[0];
  const int*   src   = (const int*)  d_in[1];
  const int*   dst   = (const int*)  d_in[2];
  const float* enorm = (const float*)d_in[3];
  const float* mu    = (const float*)d_in[4];
  const float* W1    = (const float*)d_in[5];
  const float* W2    = (const float*)d_in[6];
  float* out = (float*)d_out;

  const int E = in_sizes[1];                 // N_EDGES
  const int blocks = (E + TILE_E - 1) / TILE_E;

  // opt in to >64KB dynamic LDS (gfx1250 WGP has 320KB); ignore failure
  (void)hipFuncSetAttribute((const void*)edge_mlp_wmma,
                            hipFuncAttributeMaxDynamicSharedMemorySize,
                            LDS_BYTES);

  edge_mlp_wmma<<<blocks, 256, LDS_BYTES, stream>>>(h, src, dst, enorm, mu,
                                                    W1, W2, out, E);
}
